// InformerEncoderLayer_65386582114985
// MI455X (gfx1250) — compile-verified, matrix operand order checked by reading, not run on hardware
//
#include <hip/hip_runtime.h>
#include <hip/hip_bf16.h>
#include <stdint.h>

// Problem constants (match reference)
#define BB 8
#define LL 2048
#define DD 512
#define FFDIM 2048
#define HH 8
#define HD 64
#define KTOP 409          // int(2048 * 0.2)
#define SELPAD 448        // KTOP padded to 7 * 64
#define QTILES 7

typedef __attribute__((ext_vector_type(16))) __bf16 bf16x16;
typedef __attribute__((ext_vector_type(8)))  __bf16 bf16x8;
typedef __attribute__((ext_vector_type(8)))  float  f32x8;

__device__ __forceinline__ f32x8 wmma_bf16(bf16x16 a, bf16x16 b, f32x8 c) {
  return __builtin_amdgcn_wmma_f32_16x16x32_bf16(false, a, false, b, (short)0, c,
                                                 false, false);
}

// Build a 16-element bf16 A/B fragment from two aligned 8-element LDS chunks.
__device__ __forceinline__ bf16x16 ld_frag(const __bf16* p0, const __bf16* p1) {
  union { bf16x16 v; bf16x8 h[2]; } u;
  u.h[0] = *(const bf16x8*)p0;
  u.h[1] = *(const bf16x8*)p1;
  return u.v;
}

// Convert 8 consecutive floats to a bf16x8 (compiler packs v_cvt_pk_bf16_f32).
__device__ __forceinline__ bf16x8 cvt8(const float* __restrict__ s) {
  bf16x8 r;
#pragma unroll
  for (int i = 0; i < 8; ++i) r[i] = (__bf16)s[i];
  return r;
}

// ---------------------------------------------------------------------------
// Generic GEMM: C[M,N] = A[M,K] @ W[K,N] + bias  (optional ReLU)
// 256 threads = 8 waves (2x4), block tile 128x128, K-step 32 (bf16 WMMA).
// ---------------------------------------------------------------------------
template<int RELU>
__global__ __launch_bounds__(256)
void gemm_bias_kernel(const float* __restrict__ A, const float* __restrict__ W,
                      const float* __restrict__ bias, float* __restrict__ C,
                      int M, int N, int K) {
  (void)M;
  __shared__ __attribute__((aligned(16))) __bf16 As[128][40];  // [m][k], pad 8
  __shared__ __attribute__((aligned(16))) __bf16 Bs[128][40];  // [n][k] (W transposed)
  const int tid  = threadIdx.x;
  const int lane = tid & 31, wave = tid >> 5;
  const int lr = lane & 15, g = lane >> 4;
  const int wm = wave >> 2, wn = wave & 3;        // 2x4 wave grid
  const int m0 = blockIdx.y * 128, n0 = blockIdx.x * 128;

  f32x8 acc[4][2] = {};

  const int arow = tid >> 1, akb = (tid & 1) * 16;     // A tile load map
  const int bkk  = tid >> 3, bnb = (tid & 7) * 16;     // W tile load map

  for (int k0 = 0; k0 < K; k0 += 32) {
    const float* asrc = A + (size_t)(m0 + arow) * K + k0 + akb;
    const float* bsrc = W + (size_t)(k0 + bkk) * N + n0 + bnb;
    if (k0 + 32 < K) {
      __builtin_prefetch(asrc + 32, 0, 0);
      __builtin_prefetch(bsrc + (size_t)32 * N, 0, 0);
    }
    // A tile: contiguous, 16B-aligned -> two ds_store_b128 per thread.
    *(bf16x8*)&As[arow][akb]     = cvt8(asrc);
    *(bf16x8*)&As[arow][akb + 8] = cvt8(asrc + 8);
    // W tile: transposed scatter (must stay scalar).
#pragma unroll
    for (int i = 0; i < 16; ++i) Bs[bnb + i][bkk] = (__bf16)bsrc[i];
    __syncthreads();

    bf16x16 af[4], bfr[2];
#pragma unroll
    for (int fm = 0; fm < 4; ++fm) {
      int m = wm * 64 + fm * 16 + lr;
      af[fm] = ld_frag(&As[m][g * 8], &As[m][16 + g * 8]);
    }
#pragma unroll
    for (int fn = 0; fn < 2; ++fn) {
      int n = wn * 32 + fn * 16 + lr;
      bfr[fn] = ld_frag(&Bs[n][g * 8], &Bs[n][16 + g * 8]);
    }
#pragma unroll
    for (int fm = 0; fm < 4; ++fm)
#pragma unroll
      for (int fn = 0; fn < 2; ++fn)
        acc[fm][fn] = wmma_bf16(af[fm], bfr[fn], acc[fm][fn]);
    __syncthreads();
  }

#pragma unroll
  for (int fm = 0; fm < 4; ++fm)
#pragma unroll
    for (int fn = 0; fn < 2; ++fn) {
      int col = n0 + wn * 32 + fn * 16 + lr;
      float bv = bias[col];
#pragma unroll
      for (int r = 0; r < 8; ++r) {
        int row = m0 + wm * 64 + fm * 16 + r + 8 * g;
        float v = acc[fm][fn][r] + bv;
        if (RELU) v = fmaxf(v, 0.0f);
        C[(size_t)row * N + col] = v;
      }
    }
}

// ---------------------------------------------------------------------------
// Per-(b,h,l) squared L2 norm of Q head slices.
// ---------------------------------------------------------------------------
__global__ void qnorm_kernel(const float* __restrict__ q, float* __restrict__ qn) {
  int idx = blockIdx.x * blockDim.x + threadIdx.x;
  if (idx >= BB * HH * LL) return;
  int l = idx % LL;
  int bh = idx / LL;
  int b = bh / HH, h = bh % HH;
  const float* p = q + ((size_t)(b * LL + l)) * DD + h * HD;
  float s = 0.f;
#pragma unroll
  for (int d = 0; d < HD; ++d) s += p[d] * p[d];
  qn[idx] = s;  // layout [b][h][l]
}

// ---------------------------------------------------------------------------
// Per-(b,h) exact top-k via bitonic sort of (normbits<<32)|(~idx) keys.
// Ties resolved toward smaller index (matches jax.lax.top_k).
// ---------------------------------------------------------------------------
__global__ __launch_bounds__(256)
void topk_kernel(const float* __restrict__ qn, int* __restrict__ sel) {
  const int bh = blockIdx.x;
  __shared__ unsigned long long key[LL];
  for (int i = threadIdx.x; i < LL; i += 256) {
    unsigned nb = __float_as_uint(qn[(size_t)bh * LL + i]);  // norms >= 0
    key[i] = ((unsigned long long)nb << 32) | (unsigned)(0xFFFFFFFFu - (unsigned)i);
  }
  __syncthreads();
  for (int k2 = 2; k2 <= LL; k2 <<= 1) {
    for (int j = k2 >> 1; j > 0; j >>= 1) {
      for (int i = threadIdx.x; i < LL; i += 256) {
        int ixj = i ^ j;
        if (ixj > i) {
          bool up = ((i & k2) == 0);  // ascending overall
          unsigned long long a = key[i], c = key[ixj];
          if (up ? (a > c) : (a < c)) { key[i] = c; key[ixj] = a; }
        }
      }
      __syncthreads();
    }
  }
  // Largest KTOP keys are at the tail; pad list by repeating the last selected.
  for (int i = threadIdx.x; i < SELPAD; i += 256) {
    int src = (i < KTOP) ? (LL - 1 - i) : (LL - KTOP);
    sel[bh * SELPAD + i] =
        (int)(0xFFFFFFFFu - (unsigned)(key[src] & 0xFFFFFFFFull));
  }
}

// ---------------------------------------------------------------------------
// Per-(b,h,d) mean of V over L (uniform-softmax output for unselected rows).
// ---------------------------------------------------------------------------
__global__ void vmean_kernel(const float* __restrict__ v, float* __restrict__ vm) {
  int bh = blockIdx.x;
  int b = bh / HH, h = bh % HH;
  int d = threadIdx.x;
  float s = 0.f;
  for (int l = 0; l < LL; ++l) s += v[((size_t)(b * LL + l)) * DD + h * HD + d];
  vm[bh * HD + d] = s * (1.0f / LL);
}

__global__ void fill_uniform_kernel(const float* __restrict__ vm,
                                    float* __restrict__ apre) {
  int idx = blockIdx.x * blockDim.x + threadIdx.x;
  if (idx >= BB * LL * DD) return;
  int c = idx % DD;
  int rl = idx / DD;
  int b = rl / LL;
  int h = c / HD, d = c % HD;
  apre[idx] = vm[(b * HH + h) * HD + d];
}

// ---------------------------------------------------------------------------
// Flash attention over selected queries only. Block = 4 waves (128 thr),
// grid = (QTILES, B*H). Each wave owns 16 query rows; Bc = 64 keys/iter.
// ---------------------------------------------------------------------------
__global__ __launch_bounds__(128)
void flash_attn_kernel(const float* __restrict__ q, const float* __restrict__ k,
                       const float* __restrict__ v, const int* __restrict__ sel,
                       float* __restrict__ out) {
  const int bh = blockIdx.y;
  const int b = bh / HH, h = bh % HH;
  const int qt = blockIdx.x;
  const int tid = threadIdx.x;
  const int wave = tid >> 5, lane = tid & 31;
  const int lr = lane & 15, g = lane >> 4;

  __shared__ __attribute__((aligned(16))) __bf16 Kt[64][64];      // [key][hd]
  __shared__ __attribute__((aligned(16))) __bf16 Vt[64][64];      // [hd][key]
  __shared__ __attribute__((aligned(16))) __bf16 Pt[4][16][64];   // per-wave P

  // Load Q A-fragments (16 rows x 64 hd => two 16x32 frags), pre-scaled.
  const int qsel0 = sel[bh * SELPAD + qt * 64 + wave * 16 + lr];
  const float* qp = q + ((size_t)(b * LL + qsel0)) * DD + h * HD;
  bf16x16 qa[2];
#pragma unroll
  for (int c = 0; c < 2; ++c) {
    union { bf16x16 v; __bf16 e[16]; } u;
#pragma unroll
    for (int j = 0; j < 16; ++j) {
      int kk = c * 32 + ((j < 8) ? (g * 8 + j) : (16 + g * 8 + (j - 8)));
      u.e[j] = (__bf16)(qp[kk] * 0.125f);  // scale = hd^-0.5
    }
    qa[c] = u.v;
  }

  f32x8 accO[4] = {};
  float rowmax[8], rowsum[8];
#pragma unroll
  for (int r = 0; r < 8; ++r) { rowmax[r] = -3.0e38f; rowsum[r] = 0.f; }

  const int ldkey = tid >> 1;
  const int ldhd0 = (tid & 1) * 32;

  for (int kt = 0; kt < LL / 64; ++kt) {
    const float* kp = k + ((size_t)(b * LL + kt * 64 + ldkey)) * DD + h * HD + ldhd0;
    const float* vp = v + ((size_t)(b * LL + kt * 64 + ldkey)) * DD + h * HD + ldhd0;
    // K tile rows are contiguous & 16B-aligned -> four ds_store_b128.
#pragma unroll
    for (int c4 = 0; c4 < 4; ++c4)
      *(bf16x8*)&Kt[ldkey][ldhd0 + c4 * 8] = cvt8(kp + c4 * 8);
    // V tile is a transpose scatter (must stay scalar).
#pragma unroll
    for (int i = 0; i < 32; ++i) Vt[ldhd0 + i][ldkey] = (__bf16)vp[i];
    __syncthreads();

    // S = Q @ K^T  (contraction over hd = 2 WMMA per 16-col fragment)
    f32x8 accS[4];
#pragma unroll
    for (int fn = 0; fn < 4; ++fn) {
      int n = fn * 16 + lr;  // key column
      f32x8 s = {};
      s = wmma_bf16(qa[0], ld_frag(&Kt[n][g * 8], &Kt[n][16 + g * 8]), s);
      s = wmma_bf16(qa[1], ld_frag(&Kt[n][32 + g * 8], &Kt[n][48 + g * 8]), s);
      accS[fn] = s;
    }

    // Online softmax: rows live across the 16-lane group (n = lane%16).
#pragma unroll
    for (int r = 0; r < 8; ++r) {
      float mx = fmaxf(fmaxf(accS[0][r], accS[1][r]), fmaxf(accS[2][r], accS[3][r]));
#pragma unroll
      for (int m = 8; m >= 1; m >>= 1) mx = fmaxf(mx, __shfl_xor(mx, m, 32));
      float mnew = fmaxf(rowmax[r], mx);
      float alpha = __expf(rowmax[r] - mnew);
      rowmax[r] = mnew;
      float ps = 0.f;
#pragma unroll
      for (int fn = 0; fn < 4; ++fn) {
        float p = __expf(accS[fn][r] - mnew);
        accS[fn][r] = p;
        ps += p;
      }
#pragma unroll
      for (int m = 8; m >= 1; m >>= 1) ps += __shfl_xor(ps, m, 32);
      rowsum[r] = rowsum[r] * alpha + ps;
#pragma unroll
      for (int fn = 0; fn < 4; ++fn) accO[fn][r] *= alpha;
    }

    // C-layout P -> LDS (wave-private) -> A-layout fragments for P @ V.
#pragma unroll
    for (int fn = 0; fn < 4; ++fn)
#pragma unroll
      for (int r = 0; r < 8; ++r)
        Pt[wave][r + 8 * g][fn * 16 + lr] = (__bf16)accS[fn][r];

    bf16x16 pa0 = ld_frag(&Pt[wave][lr][g * 8],      &Pt[wave][lr][16 + g * 8]);
    bf16x16 pa1 = ld_frag(&Pt[wave][lr][32 + g * 8], &Pt[wave][lr][48 + g * 8]);

#pragma unroll
    for (int fn = 0; fn < 4; ++fn) {
      int n = fn * 16 + lr;  // hd column
      accO[fn] = wmma_bf16(pa0, ld_frag(&Vt[n][g * 8], &Vt[n][16 + g * 8]), accO[fn]);
      accO[fn] = wmma_bf16(pa1, ld_frag(&Vt[n][32 + g * 8], &Vt[n][48 + g * 8]), accO[fn]);
    }
    __syncthreads();
  }

  // Normalize and scatter back to the selected rows.
#pragma unroll
  for (int r = 0; r < 8; ++r) {
    int mrow = r + 8 * g;
    int srow = sel[bh * SELPAD + qt * 64 + wave * 16 + mrow];
    float inv = 1.0f / rowsum[r];
    float* op = out + ((size_t)(b * LL + srow)) * DD + h * HD;
#pragma unroll
    for (int fn = 0; fn < 4; ++fn) op[fn * 16 + lr] = accO[fn][r] * inv;
  }
}

// ---------------------------------------------------------------------------
// out = LayerNorm(x + y) [* gamma + beta], optionally + res (fused residual).
// One 128-thread block per row (D = 512).
// ---------------------------------------------------------------------------
template<int ADDRES>
__global__ __launch_bounds__(128)
void add_ln_kernel(const float* __restrict__ x, const float* __restrict__ y,
                   const float* __restrict__ gam, const float* __restrict__ bet,
                   const float* __restrict__ res, float* __restrict__ out) {
  int row = blockIdx.x;
  const float* xr = x + (size_t)row * DD;
  const float* yr = y + (size_t)row * DD;
  float vals[4];
  float s = 0.f, s2 = 0.f;
#pragma unroll
  for (int i = 0; i < 4; ++i) {
    int c = threadIdx.x + i * 128;
    float v = xr[c] + yr[c];
    vals[i] = v;
    s += v;
    s2 += v * v;
  }
#pragma unroll
  for (int m = 16; m >= 1; m >>= 1) {
    s  += __shfl_xor(s, m, 32);
    s2 += __shfl_xor(s2, m, 32);
  }
  __shared__ float sb[8];
  int wave = threadIdx.x >> 5, lane = threadIdx.x & 31;
  if (lane == 0) { sb[wave] = s; sb[4 + wave] = s2; }
  __syncthreads();
  s  = sb[0] + sb[1] + sb[2] + sb[3];
  s2 = sb[4] + sb[5] + sb[6] + sb[7];
  float mu = s * (1.0f / DD);
  float var = s2 * (1.0f / DD) - mu * mu;
  float rstd = rsqrtf(var + 1e-5f);
#pragma unroll
  for (int i = 0; i < 4; ++i) {
    int c = threadIdx.x + i * 128;
    float v = (vals[i] - mu) * rstd * gam[c] + bet[c];
    if (ADDRES) v += res[(size_t)row * DD + c];
    out[(size_t)row * DD + c] = v;
  }
}

// ---------------------------------------------------------------------------
// conv1d(k=3, pad=1) over L, channels D->D, as 3 accumulated WMMA GEMM taps.
// X is fused (B*L, D) row-major; W_t[k][n] = conv_w[n][k][t]. ReLU fused.
// ---------------------------------------------------------------------------
__global__ __launch_bounds__(256)
void conv_gemm_kernel(const float* __restrict__ X, const float* __restrict__ Wc,
                      const float* __restrict__ bias, float* __restrict__ Y) {
  __shared__ __attribute__((aligned(16))) __bf16 As[128][40];
  __shared__ __attribute__((aligned(16))) __bf16 Bs[128][40];
  const int tid  = threadIdx.x;
  const int lane = tid & 31, wave = tid >> 5;
  const int lr = lane & 15, g = lane >> 4;
  const int wm = wave >> 2, wn = wave & 3;
  const int m0 = blockIdx.y * 128, n0 = blockIdx.x * 128;

  f32x8 acc[4][2] = {};

  const int arow = tid >> 1, akb = (tid & 1) * 16;
  const int bkk  = tid >> 3, bnb = (tid & 7) * 16;

  for (int step = 0; step < 48; ++step) {            // 3 taps x (512/32)
    const int t  = step / 16;
    const int k0 = (step % 16) * 32;
    {
      int gr = m0 + arow;
      int l  = gr & (LL - 1);                        // tiles never straddle batches
      int ls = l + t - 1;
      bool valid = (ls >= 0) && (ls < LL);
      const float* asrc = X + (size_t)(gr + t - 1) * DD + k0 + akb;
      bf16x8 z = {};
      *(bf16x8*)&As[arow][akb]     = valid ? cvt8(asrc)     : z;
      *(bf16x8*)&As[arow][akb + 8] = valid ? cvt8(asrc + 8) : z;
    }
#pragma unroll
    for (int i = 0; i < 16; ++i)
      Bs[bnb + i][bkk] =
          (__bf16)Wc[((size_t)(n0 + bnb + i) * DD + (k0 + bkk)) * 3 + t];
    __syncthreads();

    bf16x16 af[4], bfr[2];
#pragma unroll
    for (int fm = 0; fm < 4; ++fm) {
      int m = wm * 64 + fm * 16 + lr;
      af[fm] = ld_frag(&As[m][g * 8], &As[m][16 + g * 8]);
    }
#pragma unroll
    for (int fn = 0; fn < 2; ++fn) {
      int n = wn * 32 + fn * 16 + lr;
      bfr[fn] = ld_frag(&Bs[n][g * 8], &Bs[n][16 + g * 8]);
    }
#pragma unroll
    for (int fm = 0; fm < 4; ++fm)
#pragma unroll
      for (int fn = 0; fn < 2; ++fn)
        acc[fm][fn] = wmma_bf16(af[fm], bfr[fn], acc[fm][fn]);
    __syncthreads();
  }

#pragma unroll
  for (int fm = 0; fm < 4; ++fm)
#pragma unroll
    for (int fn = 0; fn < 2; ++fn) {
      int col = n0 + wn * 32 + fn * 16 + lr;
      float bv = bias[col];
#pragma unroll
      for (int r = 0; r < 8; ++r) {
        int row = m0 + wm * 64 + fm * 16 + r + 8 * g;
        Y[(size_t)row * DD + col] = fmaxf(acc[fm][fn][r] + bv, 0.0f);
      }
    }
}

__global__ void maxpool_kernel(const float* __restrict__ yc, float* __restrict__ out) {
  int idx = blockIdx.x * blockDim.x + threadIdx.x;
  if (idx >= BB * (LL / 2) * DD) return;
  int c = idx % DD;
  int r = idx / DD;
  int lo = r % (LL / 2);
  int b = r / (LL / 2);
  size_t base = ((size_t)(b * LL + 2 * lo)) * DD + c;
  out[idx] = fmaxf(yc[base], yc[base + DD]);
}

// ---------------------------------------------------------------------------
extern "C" void kernel_launch(void* const* d_in, const int* in_sizes, int n_in,
                              void* d_out, int out_size, void* d_ws, size_t ws_size,
                              hipStream_t stream) {
  (void)in_sizes; (void)n_in; (void)out_size; (void)ws_size;
  const float* x     = (const float*)d_in[0];
  const float* q_w   = (const float*)d_in[1];
  const float* q_b   = (const float*)d_in[2];
  const float* k_w   = (const float*)d_in[3];
  const float* k_b   = (const float*)d_in[4];
  const float* v_w   = (const float*)d_in[5];
  const float* v_b   = (const float*)d_in[6];
  const float* o_w   = (const float*)d_in[7];
  const float* o_b   = (const float*)d_in[8];
  const float* ln1_g = (const float*)d_in[9];
  const float* ln1_b = (const float*)d_in[10];
  const float* f1_w  = (const float*)d_in[11];
  const float* f1_b  = (const float*)d_in[12];
  const float* f2_w  = (const float*)d_in[13];
  const float* f2_b  = (const float*)d_in[14];
  const float* ln2_g = (const float*)d_in[15];
  const float* ln2_b = (const float*)d_in[16];
  const float* cw    = (const float*)d_in[17];
  const float* cb    = (const float*)d_in[18];
  float* out = (float*)d_out;

  float* ws = (float*)d_ws;
  const size_t SZ = (size_t)BB * LL * DD;     // 8,388,608 floats
  float* qbuf = ws;
  float* kbuf = qbuf + SZ;
  float* vbuf = kbuf + SZ;
  float* apre = vbuf + SZ;                    // pre-projection attention output
  float* aout = apre + SZ;                    // attn_out
  float* h1   = aout + SZ;
  float* ff1  = h1 + SZ;                      // B*L*F floats
  float* qn   = ff1 + (size_t)BB * LL * FFDIM;
  int*   sel  = (int*)(qn + (size_t)BB * HH * LL);
  float* vm   = (float*)(sel + BB * HH * SELPAD);
  float* ffo   = apre;   // reuse: apre dead after O-projection
  float* fused = qbuf;   // reuse: q dead after attention
  float* yconv = kbuf;   // reuse: k dead after attention

  const int M = BB * LL;
  dim3 blk(256);
  dim3 g512(DD / 128, M / 128);
  dim3 g2048(FFDIM / 128, M / 128);

  // QKV projections
  gemm_bias_kernel<0><<<g512, blk, 0, stream>>>(x, q_w, q_b, qbuf, M, DD, DD);
  gemm_bias_kernel<0><<<g512, blk, 0, stream>>>(x, k_w, k_b, kbuf, M, DD, DD);
  gemm_bias_kernel<0><<<g512, blk, 0, stream>>>(x, v_w, v_b, vbuf, M, DD, DD);

  // Top-k query selection + uniform (mean-V) default rows
  qnorm_kernel<<<(BB * HH * LL + 255) / 256, 256, 0, stream>>>(qbuf, qn);
  topk_kernel<<<BB * HH, 256, 0, stream>>>(qn, sel);
  vmean_kernel<<<BB * HH, HD, 0, stream>>>(vbuf, vm);
  fill_uniform_kernel<<<(int)(SZ / 256), 256, 0, stream>>>(vm, apre);

  // Flash attention over selected rows
  flash_attn_kernel<<<dim3(QTILES, BB * HH), 128, 0, stream>>>(qbuf, kbuf, vbuf,
                                                               sel, apre);

  // Output projection, residual+LN, FFN, residual+LN (+attn residual)
  gemm_bias_kernel<0><<<g512, blk, 0, stream>>>(apre, o_w, o_b, aout, M, DD, DD);
  add_ln_kernel<0><<<M, 128, 0, stream>>>(x, aout, ln1_g, ln1_b, nullptr, h1);
  gemm_bias_kernel<1><<<g2048, blk, 0, stream>>>(h1, f1_w, f1_b, ff1, M, FFDIM, DD);
  gemm_bias_kernel<0><<<g512, blk, 0, stream>>>(ff1, f2_w, f2_b, ffo, M, DD, FFDIM);
  add_ln_kernel<1><<<M, 128, 0, stream>>>(h1, ffo, ln2_g, ln2_b, aout, fused);

  // Distill block: conv1d(3, pad=1) + ReLU (WMMA taps), then maxpool(2,2)
  conv_gemm_kernel<<<g512, blk, 0, stream>>>(fused, cw, cb, yconv);
  maxpool_kernel<<<(BB * (LL / 2) * DD + 255) / 256, 256, 0, stream>>>(yconv, out);
}
